// WritePath_21371757265420
// MI455X (gfx1250) — matrix operand order repaired and checked by reading, not compiled, not run on hardware
//
#include <hip/hip_runtime.h>
#include <hip/hip_bf16.h>
#include <float.h>
#include <math.h>

typedef __attribute__((ext_vector_type(16))) __bf16 v16bf;
typedef __attribute__((ext_vector_type(8)))  __bf16 v8bf;
typedef __attribute__((ext_vector_type(8)))  float  v8f;
typedef unsigned int u32x4 __attribute__((ext_vector_type(4)));
typedef unsigned int u32x8 __attribute__((ext_vector_type(8)));

#define EPSF 1e-8f

constexpr int BM = 128;   // block M tile
constexpr int BN = 64;    // block N tile
constexpr int BK = 32;    // K step (matches wmma 16x16x32 K)

// ---------------------------------------------------------------------------
// helpers
// ---------------------------------------------------------------------------
__device__ __forceinline__ float block_sum256(float v, float* sbuf) {
#pragma unroll
  for (int m = 16; m >= 1; m >>= 1) v += __shfl_xor(v, m, 32);
  const int w = threadIdx.x >> 5;
  if ((threadIdx.x & 31) == 0) sbuf[w] = v;
  __syncthreads();
  float r = 0.f;
#pragma unroll
  for (int i = 0; i < 8; ++i) r += sbuf[i];
  __syncthreads();
  return r;
}

// build a 16-element bf16 fragment from two runs of 8 consecutive f32 in LDS
__device__ __forceinline__ v16bf cvt_frag16(const float* p0, const float* p1) {
  const float4 x0 = *(const float4*)p0, x1 = *(const float4*)(p0 + 4);
  const float4 y0 = *(const float4*)p1, y1 = *(const float4*)(p1 + 4);
  v16bf o;
  o[0]  = (__bf16)x0.x; o[1]  = (__bf16)x0.y; o[2]  = (__bf16)x0.z; o[3]  = (__bf16)x0.w;
  o[4]  = (__bf16)x1.x; o[5]  = (__bf16)x1.y; o[6]  = (__bf16)x1.z; o[7]  = (__bf16)x1.w;
  o[8]  = (__bf16)y0.x; o[9]  = (__bf16)y0.y; o[10] = (__bf16)y0.z; o[11] = (__bf16)y0.w;
  o[12] = (__bf16)y1.x; o[13] = (__bf16)y1.y; o[14] = (__bf16)y1.z; o[15] = (__bf16)y1.w;
  return o;
}

// ---------------------------------------------------------------------------
// Tensor Data Mover: 2D tile load global -> LDS (row-major tensor).
// D# per CDNA5 ISA. Groups 2/3 zero (2D tensor).
// dsize_code: 0=1B, 1=2B, 2=4B.  Issue from ONE wave only; TENSORcnt tracked.
// ---------------------------------------------------------------------------
__device__ __forceinline__ void tdm_load_2d(unsigned lds_addr, const void* gptr,
                                            unsigned dim0, unsigned dim1,
                                            unsigned tile0, unsigned tile1,
                                            unsigned stride0,
                                            unsigned dsize_code) {
  const unsigned long long ga = (unsigned long long)(size_t)gptr;
  u32x4 g0;
  g0[0] = 1u;                                            // count=1 (valid)
  g0[1] = lds_addr;                                      // lds_addr[31:0]
  g0[2] = (unsigned)ga;                                  // global_addr[31:0]
  g0[3] = (unsigned)((ga >> 32) & 0x01ffffffu) | (2u << 30);  // addr[56:32]|type=2
  u32x8 g1;
  g1[0] = (dsize_code << 16);                            // data_size
  g1[1] = (dim0 & 0xffffu) << 16;                        // tensor_dim0[15:0]
  g1[2] = ((dim0 >> 16) & 0xffffu) | ((dim1 & 0xffffu) << 16);
  g1[3] = ((dim1 >> 16) & 0xffffu) | (tile0 << 16);      // tile_dim0
  g1[4] = (tile1 & 0xffffu);                             // tile_dim1, tile_dim2=0
  g1[5] = stride0;                                       // dim0_stride[31:0]
  g1[6] = 0u;                                            // dim0_stride[47:32]
  g1[7] = 0u;
  u32x4 gz = {0u, 0u, 0u, 0u};
  asm volatile("tensor_load_to_lds %0, %1, %2, %3"
               :: "s"(g0), "s"(g1), "s"(gz), "s"(gz)
               : "memory");
}

// ---------------------------------------------------------------------------
// Fused projection GEMM: one launch computes all three hidden projections so
// the 128MB `hidden` tensor is streamed once (L2 = 192MB captures the reuse
// across the 20 column-blocks).  blockIdx.y selects {W_obs, W_g1, W_p1}.
// Tiles staged by the Tensor Data Mover (f32, data_size=4B) into double-
// buffered LDS; wave 0 issues the next tile's DMA, s_wait_tensorcnt(2) covers
// the current buffer, so DMA overlaps the WMMAs.  f32->bf16 conversion happens
// at fragment assembly (co-executes with XDL WMMAs).
// Block: 256 thr = 8 waves, 4(M) x 2(N); wave tile 32x32 = 2x2 wmma tiles.
// ---------------------------------------------------------------------------
__global__ __launch_bounds__(256) void k_gemm_fused(
    const float* __restrict__ A, const float* __restrict__ W_obs,
    const float* __restrict__ W_g1, const float* __restrict__ W_p1,
    float* __restrict__ obs, float* __restrict__ gh, float* __restrict__ ph,
    int M, int H) {
  __shared__ alignas(128) float sA[2][BM][BK];   // 32 KB
  __shared__ alignas(128) float sB[2][BN][BK];   // 16 KB

  const int tid = threadIdx.x;
  const int lane = tid & 31, wave = tid >> 5;
  const int wm = wave >> 1, wn = wave & 1;
  const int bM = blockIdx.x * BM;
  const int by = blockIdx.y;
  const int half = lane >> 4, l16 = lane & 15;

  const float* W;
  float* C;
  int nOff, Ncols;
  if (by < 4)       { W = W_obs; C = obs; nOff = by * BN;        Ncols = 256; }
  else if (by < 12) { W = W_g1;  C = gh;  nOff = (by - 4) * BN;  Ncols = 512; }
  else              { W = W_p1;  C = ph;  nOff = (by - 12) * BN; Ncols = 512; }

  v8f acc[2][2] = {};

  if (wave == 0) {   // prologue DMA into buffer 0
    tdm_load_2d((unsigned)(size_t)&sA[0][0][0], A + (size_t)bM * H,
                H, M, BK, BM, H, 2u);
    tdm_load_2d((unsigned)(size_t)&sB[0][0][0], W + (size_t)nOff * H,
                H, (unsigned)(Ncols), BK, BN, H, 2u);
  }

  int ib = 0;
  for (int k0 = 0; k0 < H; k0 += BK, ib ^= 1) {
    if (wave == 0) {
      const int kn = k0 + BK;
      if (kn < H) {   // DMA next tile into the other buffer, overlap compute
        tdm_load_2d((unsigned)(size_t)&sA[ib ^ 1][0][0],
                    A + (size_t)bM * H + kn, H, M, BK, BM, H, 2u);
        tdm_load_2d((unsigned)(size_t)&sB[ib ^ 1][0][0],
                    W + (size_t)nOff * H + kn, H, (unsigned)Ncols, BK, BN, H, 2u);
        __builtin_amdgcn_s_wait_tensorcnt(2);  // current buffer complete
      } else {
        __builtin_amdgcn_s_wait_tensorcnt(0);
      }
    }
    __syncthreads();

    // B fragments once per K-step (shared across the i loop)
    v16bf bfrag[2];
#pragma unroll
    for (int j = 0; j < 2; ++j) {
      const int nl = wn * 32 + j * 16 + l16;
      bfrag[j] = cvt_frag16(&sB[ib][nl][half * 16], &sB[ib][nl][half * 16 + 8]);
    }
#pragma unroll
    for (int i = 0; i < 2; ++i) {
      const int row = wm * 32 + i * 16 + l16;
      const v16bf a = cvt_frag16(&sA[ib][row][half * 16],
                                 &sA[ib][row][half * 16 + 8]);
#pragma unroll
      for (int j = 0; j < 2; ++j)
        acc[i][j] = __builtin_amdgcn_wmma_f32_16x16x32_bf16(
            false, a, false, bfrag[j], (short)0, acc[i][j], false, false);
    }
    __syncthreads();   // buffer ib free before TDM reuses it next+1 iteration
  }

  // store: C/D layout: VGPR r -> M = r + 8*half, lane%16 -> N
#pragma unroll
  for (int i = 0; i < 2; ++i)
#pragma unroll
    for (int j = 0; j < 2; ++j)
#pragma unroll
      for (int r = 0; r < 8; ++r) {
        const int m = bM + wm * 32 + i * 16 + half * 8 + r;
        const int n = nOff + wn * 32 + j * 16 + l16;
        C[(size_t)m * Ncols + n] = acc[i][j][r];
      }
}

// ---------------------------------------------------------------------------
// gate/prec second-layer + batch mean:  prec_mean[t] = mean_b gate*prec
// ---------------------------------------------------------------------------
__global__ __launch_bounds__(256) void k_gate_prec(
    const float* __restrict__ gh, const float* __restrict__ ph,
    const float* __restrict__ b_g1, const float* __restrict__ W_g2,
    const float* __restrict__ b_g2, const float* __restrict__ b_p1,
    const float* __restrict__ W_p2, const float* __restrict__ b_p2,
    float* __restrict__ prec_mean, int T, int Kd, int Bb) {
  __shared__ float sbuf[8];
  const int t = blockIdx.x;
  float s = 0.f;
  for (int b = 0; b < Bb; ++b) {
    const size_t row = (size_t)b * T + t;
    float gp = 0.f, pp = 0.f;
    for (int k = threadIdx.x; k < Kd; k += 256) {
      gp += fmaxf(gh[row * Kd + k] + b_g1[k], 0.f) * W_g2[k];
      pp += fmaxf(ph[row * Kd + k] + b_p1[k], 0.f) * W_p2[k];
    }
    const float g = block_sum256(gp, sbuf);
    const float p = block_sum256(pp, sbuf);
    const float gate = 1.f / (1.f + expf(-(g + b_g2[0])));
    const float x = p + b_p2[0];
    const float prec = (x > 20.f) ? x : log1pf(expf(x));
    s += gate * prec;
  }
  if (threadIdx.x == 0) prec_mean[t] = s / (float)Bb;
}

// ---------------------------------------------------------------------------
// obs mean over B, normalize, scale by prec_mean -> obs_beliefs (output 0),
// also emit obs_r[t] and bf16 obs_ang for the sims GEMM.  blockDim == D == 256
// ---------------------------------------------------------------------------
__global__ __launch_bounds__(256) void k_obs_post(
    const float* __restrict__ obs, const float* __restrict__ prec_mean,
    float* __restrict__ out_beliefs, float* __restrict__ obs_r,
    __bf16* __restrict__ obs_ang, int T, int D, int Bb) {
  __shared__ float sbuf[8];
  const int t = blockIdx.x, d = threadIdx.x;
  float mean = 0.f;
  for (int b = 0; b < Bb; ++b) mean += obs[((size_t)b * T + t) * D + d];
  mean *= (1.f / (float)Bb);
  const float n2 = block_sum256(mean * mean, sbuf);
  const float ang0 = mean / fmaxf(sqrtf(n2), EPSF);
  const float belief = ang0 * prec_mean[t];
  out_beliefs[(size_t)t * D + d] = belief;
  const float r2 = block_sum256(belief * belief, sbuf);
  const float r = sqrtf(r2);
  if (d == 0) obs_r[t] = r;
  obs_ang[(size_t)t * D + d] = (__bf16)(belief / fmaxf(r, EPSF));
}

// ---------------------------------------------------------------------------
// normalize belief rows -> bf16.  blockDim == D == 256
// ---------------------------------------------------------------------------
__global__ __launch_bounds__(256) void k_bel_norm(const float* __restrict__ bel,
                                                  __bf16* __restrict__ bel_b,
                                                  int N, int D) {
  __shared__ float sbuf[8];
  const int n = blockIdx.x, d = threadIdx.x;
  const float x = bel[(size_t)n * D + d];
  const float r = fmaxf(sqrtf(block_sum256(x * x, sbuf)), EPSF);
  bel_b[(size_t)n * D + d] = (__bf16)(x / r);
}

__global__ void k_any_active(const int* __restrict__ mask, int N,
                             int* __restrict__ flag) {
  __shared__ int s;
  if (threadIdx.x == 0) s = 0;
  __syncthreads();
  int f = 0;
  for (int i = threadIdx.x; i < N; i += blockDim.x) f |= (mask[i] != 0);
  if (f) atomicOr(&s, 1);
  __syncthreads();
  if (threadIdx.x == 0) *flag = s;
}

// ---------------------------------------------------------------------------
// sims = obs_ang[T,K] * bel_ang[N,K]^T  (bf16 WMMA).  Tiles staged by the
// Tensor Data Mover into double-buffered LDS (data_size=2B); wave 0 issues
// the NEXT K-tile's DMA, then s_wait_tensorcnt(2) guarantees the current
// buffer's two loads (in-order) are complete, so DMA overlaps the WMMAs.
// Fused masked per-row max/argmax over this block's 64-column strip.
// ---------------------------------------------------------------------------
__global__ __launch_bounds__(256) void k_sims(
    const __bf16* __restrict__ A, const __bf16* __restrict__ Bm,
    const int* __restrict__ mask, float* __restrict__ pval,
    int* __restrict__ pidx, int M, int Kdim, int N, int slots) {
  __shared__ alignas(128) __bf16 sA[2][BM][BK];   // 16 KB
  __shared__ alignas(128) __bf16 sB[2][BN][BK];   //  8 KB
  const int tid = threadIdx.x;
  const int lane = tid & 31, wave = tid >> 5;
  const int wm = wave >> 1, wn = wave & 1;
  const int bM = blockIdx.x * BM, bN = blockIdx.y * BN;
  const int half = lane >> 4, l16 = lane & 15;

  v8f acc[2][2] = {};

  if (wave == 0) {   // prologue DMA into buffer 0
    tdm_load_2d((unsigned)(size_t)&sA[0][0][0], A + (size_t)bM * Kdim,
                Kdim, M, BK, BM, Kdim, 1u);
    tdm_load_2d((unsigned)(size_t)&sB[0][0][0], Bm + (size_t)bN * Kdim,
                Kdim, N, BK, BN, Kdim, 1u);
  }

  int ib = 0;
  for (int k0 = 0; k0 < Kdim; k0 += BK, ib ^= 1) {
    if (wave == 0) {
      const int kn = k0 + BK;
      if (kn < Kdim) {   // DMA next tile into the other buffer, overlap compute
        tdm_load_2d((unsigned)(size_t)&sA[ib ^ 1][0][0],
                    A + (size_t)bM * Kdim + kn, Kdim, M, BK, BM, Kdim, 1u);
        tdm_load_2d((unsigned)(size_t)&sB[ib ^ 1][0][0],
                    Bm + (size_t)bN * Kdim + kn, Kdim, N, BK, BN, Kdim, 1u);
        __builtin_amdgcn_s_wait_tensorcnt(2);  // current buffer complete
      } else {
        __builtin_amdgcn_s_wait_tensorcnt(0);
      }
    }
    __syncthreads();

#pragma unroll
    for (int i = 0; i < 2; ++i) {
      const int row = wm * 32 + i * 16 + l16;
      union { v16bf v; v8bf h[2]; } a;
      a.h[0] = *(const v8bf*)&sA[ib][row][half * 16];
      a.h[1] = *(const v8bf*)&sA[ib][row][half * 16 + 8];
#pragma unroll
      for (int j = 0; j < 2; ++j) {
        const int nl = wn * 32 + j * 16 + l16;
        v16bf b = *(const v16bf*)&sB[ib][nl][half * 16];
        acc[i][j] = __builtin_amdgcn_wmma_f32_16x16x32_bf16(
            false, a.v, false, b, (short)0, acc[i][j], false, false);
      }
    }
    __syncthreads();   // buffer ib free before TDM reuses it next+1 iteration
  }

  // masked row-argmax across this wave's 32 columns (16-lane shuffle tourney)
#pragma unroll
  for (int i = 0; i < 2; ++i) {
    float bv[8]; int bi[8];
#pragma unroll
    for (int r = 0; r < 8; ++r) { bv[r] = -FLT_MAX; bi[r] = 0x7fffffff; }
#pragma unroll
    for (int j = 0; j < 2; ++j) {
      const int n = bN + wn * 32 + j * 16 + l16;
      const int act = (mask[n] != 0);
#pragma unroll
      for (int r = 0; r < 8; ++r) {
        float v = act ? acc[i][j][r] : -FLT_MAX;
        int ix = n;
#pragma unroll
        for (int m = 1; m < 16; m <<= 1) {   // stays within 16-lane groups
          const float ov = __shfl_xor(v, m, 32);
          const int oi = __shfl_xor(ix, m, 32);
          if (ov > v || (ov == v && oi < ix)) { v = ov; ix = oi; }
        }
        if (v > bv[r] || (v == bv[r] && ix < bi[r])) { bv[r] = v; bi[r] = ix; }
      }
    }
    if (l16 == 0) {
      const int slot = blockIdx.y * 2 + wn;
#pragma unroll
      for (int r = 0; r < 8; ++r) {
        const int m = bM + wm * 32 + i * 16 + half * 8 + r;
        pval[(size_t)m * slots + slot] = bv[r];
        pidx[(size_t)m * slots + slot] = bi[r];
      }
    }
  }
}

// ---------------------------------------------------------------------------
// final: reduce 256 partials per token, threshold, write 3 output arrays
// ---------------------------------------------------------------------------
__global__ __launch_bounds__(256) void k_final(
    const float* __restrict__ pval, const int* __restrict__ pidx,
    const float* __restrict__ obs_r, const int* __restrict__ flag,
    float* __restrict__ out_sim, float* __restrict__ out_slot,
    float* __restrict__ out_meaningful, int T, int slots) {
  __shared__ float sv[8];
  __shared__ int si[8];
  const int t = blockIdx.x;
  float v = pval[(size_t)t * slots + threadIdx.x];
  int ix = pidx[(size_t)t * slots + threadIdx.x];
#pragma unroll
  for (int m = 16; m >= 1; m >>= 1) {
    const float ov = __shfl_xor(v, m, 32);
    const int oi = __shfl_xor(ix, m, 32);
    if (ov > v || (ov == v && oi < ix)) { v = ov; ix = oi; }
  }
  const int w = threadIdx.x >> 5;
  if ((threadIdx.x & 31) == 0) { sv[w] = v; si[w] = ix; }
  __syncthreads();
  if (threadIdx.x == 0) {
    v = sv[0]; ix = si[0];
    for (int i = 1; i < 8; ++i)
      if (sv[i] > v || (sv[i] == v && si[i] < ix)) { v = sv[i]; ix = si[i]; }
    const int any = *flag;
    const float best = any ? v : 0.f;
    const int bidx = any ? ix : -1;
    const float r = obs_r[t];
    const int meaningful = (r > 0.05f);
    const int matched = meaningful && (best > 0.5f);
    out_sim[t] = matched ? best : 0.f;
    out_slot[t] = matched ? (float)bidx : -1.f;
    out_meaningful[t] = meaningful ? 1.f : 0.f;
  }
}

// ---------------------------------------------------------------------------
extern "C" void kernel_launch(void* const* d_in, const int* in_sizes, int n_in,
                              void* d_out, int out_size, void* d_ws,
                              size_t ws_size, hipStream_t stream) {
  (void)in_sizes; (void)n_in; (void)out_size; (void)ws_size;
  constexpr int Bb = 4, T = 4096, H = 2048, D = 256, Nn = 8192, Kd = 512;

  const float* hidden = (const float*)d_in[0];
  const float* beliefs = (const float*)d_in[1];
  const int*   amask  = (const int*)d_in[2];
  const float* W_obs = (const float*)d_in[3];
  const float* W_g1 = (const float*)d_in[4];
  const float* b_g1 = (const float*)d_in[5];
  const float* W_g2 = (const float*)d_in[6];
  const float* b_g2 = (const float*)d_in[7];
  const float* W_p1 = (const float*)d_in[8];
  const float* b_p1 = (const float*)d_in[9];
  const float* W_p2 = (const float*)d_in[10];
  const float* b_p2 = (const float*)d_in[11];

  char* ws = (char*)d_ws;
  size_t off = 0;
  auto alloc = [&](size_t bytes) -> void* {
    void* p = ws + off;
    off += (bytes + 255) & ~(size_t)255;
    return p;
  };
  const int M = Bb * T;
  float* obs_ws = (float*)alloc((size_t)M * D * 4);
  float* gh_ws = (float*)alloc((size_t)M * Kd * 4);
  float* ph_ws = (float*)alloc((size_t)M * Kd * 4);
  float* prec_mean = (float*)alloc((size_t)T * 4);
  float* obs_r = (float*)alloc((size_t)T * 4);
  __bf16* obs_ang = (__bf16*)alloc((size_t)T * D * 2);
  __bf16* bel_b = (__bf16*)alloc((size_t)Nn * D * 2);
  const int SLOTS = (Nn / BN) * 2;  // 256
  float* pval = (float*)alloc((size_t)T * SLOTS * 4);
  int* pidx = (int*)alloc((size_t)T * SLOTS * 4);
  int* flag = (int*)alloc(4);

  dim3 blk(256);
  // one launch for all three hidden projections: grid y = 4 (obs) + 8 + 8
  k_gemm_fused<<<dim3(M / BM, 20), blk, 0, stream>>>(
      hidden, W_obs, W_g1, W_p1, obs_ws, gh_ws, ph_ws, M, H);
  k_gate_prec<<<dim3(T), blk, 0, stream>>>(gh_ws, ph_ws, b_g1, W_g2, b_g2,
                                           b_p1, W_p2, b_p2, prec_mean, T, Kd, Bb);
  float* out = (float*)d_out;
  k_obs_post<<<dim3(T), blk, 0, stream>>>(obs_ws, prec_mean, out, obs_r,
                                          obs_ang, T, D, Bb);
  k_bel_norm<<<dim3(Nn), blk, 0, stream>>>(beliefs, bel_b, Nn, D);
  k_any_active<<<dim3(1), dim3(1024), 0, stream>>>(amask, Nn, flag);
  k_sims<<<dim3(T / BM, Nn / BN), blk, 0, stream>>>(obs_ang, bel_b, amask,
                                                    pval, pidx, T, D, Nn, SLOTS);
  k_final<<<dim3(T), blk, 0, stream>>>(pval, pidx, obs_r, flag,
                                       out + (size_t)T * D,
                                       out + (size_t)T * D + T,
                                       out + (size_t)T * D + 2 * T, T, SLOTS);
}
